// ResGCN4_58128087384884
// MI455X (gfx1250) — matrix operand order repaired and checked by reading, not compiled, not run on hardware
//
#include <hip/hip_runtime.h>
#include <hip/hip_bf16.h>

typedef __attribute__((ext_vector_type(16))) __bf16 v16bf;
typedef __attribute__((ext_vector_type(8)))  float  v8f;

#define NN 100000
#define NE 1600000

__device__ __forceinline__ unsigned int pack_bf16(float lo, float hi) {
  // round-to-nearest-even f32 -> bf16, packed pair (used only in streaming passes)
  unsigned int ul = __float_as_uint(lo);
  unsigned int uh = __float_as_uint(hi);
  ul = (ul + 0x7FFFu + ((ul >> 16) & 1u)) >> 16;
  uh = (uh + 0x7FFFu + ((uh >> 16) & 1u)) >> 16;
  return (ul & 0xFFFFu) | (uh << 16);
}

__device__ __forceinline__ float bf_lo(unsigned int p) {
  return __uint_as_float(p << 16);
}
__device__ __forceinline__ float bf_hi(unsigned int p) {
  return __uint_as_float(p & 0xFFFF0000u);
}

union BFrag { unsigned int u[8]; v16bf v; };

// ---------------------------------------------------------------------------
// Pack fp32 W[K,N] into bf16 WMMA B-fragments:
// dword index = ((kt*CT + nt)*32 + lane)*8 + v ;  k pair = kt*32 + 2v + 8*half + 8*(v>>2)
// ---------------------------------------------------------------------------
__global__ void pack_w_bf16(const float* __restrict__ W, unsigned int* __restrict__ Wp,
                            int K, int N, int CT) {
  int idx = blockIdx.x * blockDim.x + threadIdx.x;
  int total = (K >> 5) * CT * 256;
  if (idx >= total) return;
  int v    = idx & 7;
  int lane = (idx >> 3) & 31;
  int tile = idx >> 8;
  int nt = tile % CT, kt = tile / CT;
  int half = lane >> 4, r = lane & 15;
  int k0  = kt * 32 + 2 * v + 8 * half + ((v >> 2) << 3);
  int col = nt * 16 + r;
  float lo = 0.f, hi = 0.f;
  if (col < N) { lo = W[k0 * N + col]; hi = W[(k0 + 1) * N + col]; }
  Wp[idx] = pack_bf16(lo, hi);
}

// fp32 -> row-major bf16 (packed pairs), 4 elements/thread
__global__ void cvt_f32_bf16(const float4* __restrict__ in, uint2* __restrict__ out,
                             int n4) {
  int idx = blockIdx.x * blockDim.x + threadIdx.x;
  if (idx >= n4) return;
  float4 v = in[idx];
  uint2 o;
  o.x = pack_bf16(v.x, v.y);
  o.y = pack_bf16(v.z, v.w);
  out[idx] = o;
}

// ---------------------------------------------------------------------------
// Y[M,128] = Xbf[M,K](bf16) @ W(prepacked) (+ bias). One wave = 16x128 strip.
// Inner loop: 2x b128 A loads + 8x (2x b128 B loads + wmma). No conversion VALU.
// ---------------------------------------------------------------------------
template <int K, bool HAS_BIAS>
__global__ void gemm128_wmma(const unsigned short* __restrict__ Xbf,
                             const unsigned int* __restrict__ Wp,
                             const float* __restrict__ bias,
                             float* __restrict__ Y, int MT) {
  const int lane = threadIdx.x & 31;
  const int wave = threadIdx.x >> 5;
  const int mt = blockIdx.x * 8 + wave;
  if (mt >= MT) return;                      // wave-uniform
  const int half = lane >> 4, r = lane & 15;
  const int m0 = mt << 4;

  v8f c[8];
#pragma unroll
  for (int t = 0; t < 8; ++t) c[t] = (v8f){0.f,0.f,0.f,0.f,0.f,0.f,0.f,0.f};

  const uint4* xrow = (const uint4*)(Xbf + (size_t)(m0 + r) * K);  // 8 bf16 / uint4

#pragma unroll
  for (int kb = 0; kb < K; kb += 32) {
    uint4 a01 = xrow[(kb >> 3) + half];       // bf16 k = kb+8h .. +7
    uint4 a23 = xrow[(kb >> 3) + half + 2];   // bf16 k = kb+16+8h .. +7
    BFrag a;
    a.u[0] = a01.x; a.u[1] = a01.y; a.u[2] = a01.z; a.u[3] = a01.w;
    a.u[4] = a23.x; a.u[5] = a23.y; a.u[6] = a23.z; a.u[7] = a23.w;
    const int kt = kb >> 5;
#pragma unroll
    for (int nt = 0; nt < 8; ++nt) {
      const uint4* bp = (const uint4*)(Wp + (((size_t)kt * 8 + nt) * 32 + lane) * 8);
      uint4 b0 = bp[0], b1 = bp[1];
      BFrag b;
      b.u[0] = b0.x; b.u[1] = b0.y; b.u[2] = b0.z; b.u[3] = b0.w;
      b.u[4] = b1.x; b.u[5] = b1.y; b.u[6] = b1.z; b.u[7] = b1.w;
      c[nt] = __builtin_amdgcn_wmma_f32_16x16x32_bf16(false, a.v, false, b.v,
                                                      (short)0, c[nt], false, false);
    }
  }

#pragma unroll
  for (int nt = 0; nt < 8; ++nt) {
    const int col = nt * 16 + r;
    float bv = HAS_BIAS ? bias[col] : 0.f;
#pragma unroll
    for (int i = 0; i < 8; ++i)
      Y[(size_t)(m0 + i + half * 8) * 128 + col] = c[nt][i] + bv;
  }
}

// ---------------------------------------------------------------------------
// Y[M,40] = concat(X0,X1,X2)[M,384](bf16) @ W[384,40] (prepacked, padded 48).
// K fully unrolled so the concat pointer select is compile-time.
// ---------------------------------------------------------------------------
__global__ void gemm_cat3_wmma(const unsigned short* __restrict__ X0,
                               const unsigned short* __restrict__ X1,
                               const unsigned short* __restrict__ X2,
                               const unsigned int* __restrict__ Wp,
                               float* __restrict__ Y, int MT) {
  const int lane = threadIdx.x & 31;
  const int wave = threadIdx.x >> 5;
  const int mt = blockIdx.x * 8 + wave;
  if (mt >= MT) return;
  const int half = lane >> 4, r = lane & 15;
  const int m0 = mt << 4;

  v8f c[3];
#pragma unroll
  for (int t = 0; t < 3; ++t) c[t] = (v8f){0.f,0.f,0.f,0.f,0.f,0.f,0.f,0.f};

#pragma unroll
  for (int kb = 0; kb < 384; kb += 32) {
    const unsigned short* Xp = (kb < 128) ? X0 : ((kb < 256) ? X1 : X2);
    const uint4* xrow = (const uint4*)(Xp + (size_t)(m0 + r) * 128 + (kb & 127));
    uint4 a01 = xrow[half];
    uint4 a23 = xrow[half + 2];
    BFrag a;
    a.u[0] = a01.x; a.u[1] = a01.y; a.u[2] = a01.z; a.u[3] = a01.w;
    a.u[4] = a23.x; a.u[5] = a23.y; a.u[6] = a23.z; a.u[7] = a23.w;
    const int kt = kb >> 5;
#pragma unroll
    for (int nt = 0; nt < 3; ++nt) {
      const uint4* bp = (const uint4*)(Wp + (((size_t)kt * 3 + nt) * 32 + lane) * 8);
      uint4 b0 = bp[0], b1 = bp[1];
      BFrag b;
      b.u[0] = b0.x; b.u[1] = b0.y; b.u[2] = b0.z; b.u[3] = b0.w;
      b.u[4] = b1.x; b.u[5] = b1.y; b.u[6] = b1.z; b.u[7] = b1.w;
      c[nt] = __builtin_amdgcn_wmma_f32_16x16x32_bf16(false, a.v, false, b.v,
                                                      (short)0, c[nt], false, false);
    }
  }

#pragma unroll
  for (int nt = 0; nt < 3; ++nt) {
    const int col = nt * 16 + r;
    if (col < 40) {
#pragma unroll
      for (int i = 0; i < 8; ++i)
        Y[(size_t)(m0 + i + half * 8) * 40 + col] = c[nt][i];
    }
  }
}

// dst[rows[e], :] += vals[e] * src[cols[e], :]   (COO scatter, float4 chunks)
__global__ void spmm_scatter(const int* __restrict__ rows,
                             const int* __restrict__ cols,
                             const float* __restrict__ vals,
                             const float* __restrict__ src,
                             float* __restrict__ dst,
                             int nE, int chunks, int ncols) {
  unsigned int idx = blockIdx.x * blockDim.x + threadIdx.x;
  unsigned int total = (unsigned int)nE * (unsigned int)chunks;
  if (idx >= total) return;
  int e = idx / chunks;
  int cb = (idx - e * chunks) * 4;
  int sc = cols[e], dr = rows[e];
  float v = vals[e];
  const float4 s = *(const float4*)(src + (size_t)sc * ncols + cb);
  float* dp = dst + (size_t)dr * ncols + cb;
  unsafeAtomicAdd(dp + 0, s.x * v);
  unsafeAtomicAdd(dp + 1, s.y * v);
  unsafeAtomicAdd(dp + 2, s.z * v);
  unsafeAtomicAdd(dp + 3, s.w * v);
}

// out(bf16) = relu(agg + bias[col]) + res ; res is fp32 (layer1) or bf16 (2,3)
template <bool RES_BF16>
__global__ void combine_relu_bf16(const float4* __restrict__ agg,
                                  const float4* __restrict__ bias,  // 32 entries
                                  const void* __restrict__ res,
                                  uint2* __restrict__ out, int total4) {
  int idx = blockIdx.x * blockDim.x + threadIdx.x;
  if (idx >= total4) return;
  float4 b = bias[idx & 31];
  float4 a = agg[idx];
  float4 r;
  if (RES_BF16) {
    uint2 rb = ((const uint2*)res)[idx];
    r.x = bf_lo(rb.x); r.y = bf_hi(rb.x);
    r.z = bf_lo(rb.y); r.w = bf_hi(rb.y);
  } else {
    r = ((const float4*)res)[idx];
  }
  float ox = fmaxf(a.x + b.x, 0.f) + r.x;
  float oy = fmaxf(a.y + b.y, 0.f) + r.y;
  float oz = fmaxf(a.z + b.z, 0.f) + r.z;
  float ow = fmaxf(a.w + b.w, 0.f) + r.w;
  uint2 o;
  o.x = pack_bf16(ox, oy);
  o.y = pack_bf16(oz, ow);
  out[idx] = o;
}

// out[row,:] = log_softmax(agg[row,:] + bias), 40 classes, one wave per row
__global__ void logsoftmax40(const float* __restrict__ agg,
                             const float* __restrict__ bias,
                             float* __restrict__ out, int M) {
  int lane = threadIdx.x & 31;
  int row = blockIdx.x * 8 + (threadIdx.x >> 5);
  if (row >= M) return;
  const float* a = agg + (size_t)row * 40;
  float v0 = a[lane] + bias[lane];
  bool hi = (lane + 32) < 40;
  float v1 = hi ? (a[lane + 32] + bias[lane + 32]) : -3.4e38f;
  float m = fmaxf(v0, v1);
  for (int off = 16; off; off >>= 1) m = fmaxf(m, __shfl_xor(m, off, 32));
  float s = __expf(v0 - m) + (hi ? __expf(v1 - m) : 0.f);
  for (int off = 16; off; off >>= 1) s += __shfl_xor(s, off, 32);
  float lse = m + __logf(s);
  out[(size_t)row * 40 + lane] = v0 - lse;
  if (hi) out[(size_t)row * 40 + lane + 32] = v1 - lse;
}

extern "C" void kernel_launch(void* const* d_in, const int* in_sizes, int n_in,
                              void* d_out, int out_size, void* d_ws, size_t ws_size,
                              hipStream_t stream) {
  const float* x     = (const float*)d_in[0];
  const int*   arows = (const int*)d_in[1];
  const int*   acols = (const int*)d_in[2];
  const float* avals = (const float*)d_in[3];
  const float* w_res = (const float*)d_in[4];
  const float* b_res = (const float*)d_in[5];
  const float* w1 = (const float*)d_in[6];  const float* b1 = (const float*)d_in[7];
  const float* w2 = (const float*)d_in[8];  const float* b2 = (const float*)d_in[9];
  const float* w3 = (const float*)d_in[10]; const float* b3 = (const float*)d_in[11];
  const float* w4 = (const float*)d_in[12]; const float* b4 = (const float*)d_in[13];
  float* out = (float*)d_out;

  // ---- workspace layout (bytes) ----
  char* wsb = (char*)d_ws;
  const size_t S  = (size_t)NN * 128;                 // per-layer elements
  float* z    = (float*)wsb;                          // fp32 residual (layer 1)
  float* sup  = (float*)(wsb + S * 4);                // fp32 support / s4
  float* agg  = (float*)(wsb + 2 * S * 4);            // fp32 agg / agg4
  unsigned short* xbf = (unsigned short*)(wsb + 3 * S * 4);            // [NN,256] bf16
  unsigned short* x1b = (unsigned short*)(wsb + 3 * S * 4 + (size_t)NN * 512);
  unsigned short* x2b = x1b + S;                      // [NN,128] bf16 each
  unsigned short* x3b = x2b + S;
  unsigned int* pw = (unsigned int*)(x3b + S);        // packed weights
  const int szK256 = 8 * 8 * 256;
  const int szK128 = 4 * 8 * 256;
  const int szK384 = 12 * 3 * 256;
  unsigned int* pw_res = pw;
  unsigned int* pw1    = pw_res + szK256;
  unsigned int* pw2    = pw1    + szK256;
  unsigned int* pw3    = pw2    + szK128;
  unsigned int* pw4    = pw3    + szK128;

  float* s4   = sup;
  float* agg4 = agg;

  dim3 blk(256);
  const int MT = NN / 16;                        // 6250
  const int gemmBlk  = (MT + 7) / 8;
  const int spmmBlk  = (NE * 32 + 255) / 256;
  const int spmm4Blk = (NE * 10 + 255) / 256;
  const int cb4 = (int)((S / 4 + 255) / 256);

  // weight packing + input conversion
  pack_w_bf16<<<(szK256 + 255) / 256, blk, 0, stream>>>(w_res, pw_res, 256, 128, 8);
  pack_w_bf16<<<(szK256 + 255) / 256, blk, 0, stream>>>(w1,    pw1,    256, 128, 8);
  pack_w_bf16<<<(szK128 + 255) / 256, blk, 0, stream>>>(w2,    pw2,    128, 128, 8);
  pack_w_bf16<<<(szK128 + 255) / 256, blk, 0, stream>>>(w3,    pw3,    128, 128, 8);
  pack_w_bf16<<<(szK384 + 255) / 256, blk, 0, stream>>>(w4,    pw4,    384,  40, 3);
  const int xn4 = NN * 256 / 4;
  cvt_f32_bf16<<<(xn4 + 255) / 256, blk, 0, stream>>>((const float4*)x, (uint2*)xbf, xn4);

  // z = x @ w_res + b_res ; sup = x @ w1
  gemm128_wmma<256, true ><<<gemmBlk, blk, 0, stream>>>(xbf, pw_res, b_res, z, MT);
  gemm128_wmma<256, false><<<gemmBlk, blk, 0, stream>>>(xbf, pw1, nullptr, sup, MT);
  hipMemsetAsync(agg, 0, S * sizeof(float), stream);
  spmm_scatter<<<spmmBlk, blk, 0, stream>>>(arows, acols, avals, sup, agg, NE, 32, 128);
  combine_relu_bf16<false><<<cb4, blk, 0, stream>>>((const float4*)agg, (const float4*)b1,
                                                    (const void*)z, (uint2*)x1b, (int)(S / 4));
  // layer 2
  gemm128_wmma<128, false><<<gemmBlk, blk, 0, stream>>>(x1b, pw2, nullptr, sup, MT);
  hipMemsetAsync(agg, 0, S * sizeof(float), stream);
  spmm_scatter<<<spmmBlk, blk, 0, stream>>>(arows, acols, avals, sup, agg, NE, 32, 128);
  combine_relu_bf16<true><<<cb4, blk, 0, stream>>>((const float4*)agg, (const float4*)b2,
                                                   (const void*)x1b, (uint2*)x2b, (int)(S / 4));
  // layer 3
  gemm128_wmma<128, false><<<gemmBlk, blk, 0, stream>>>(x2b, pw3, nullptr, sup, MT);
  hipMemsetAsync(agg, 0, S * sizeof(float), stream);
  spmm_scatter<<<spmmBlk, blk, 0, stream>>>(arows, acols, avals, sup, agg, NE, 32, 128);
  combine_relu_bf16<true><<<cb4, blk, 0, stream>>>((const float4*)agg, (const float4*)b3,
                                                   (const void*)x2b, (uint2*)x3b, (int)(S / 4));
  // final: s4 = concat(x3,x2,x1) @ w4 ; agg4 = adj @ s4 ; out = log_softmax(agg4+b4)
  gemm_cat3_wmma<<<gemmBlk, blk, 0, stream>>>(x3b, x2b, x1b, pw4, s4, MT);
  hipMemsetAsync(agg4, 0, (size_t)NN * 40 * sizeof(float), stream);
  spmm_scatter<<<spmm4Blk, blk, 0, stream>>>(arows, acols, avals, s4, agg4, NE, 10, 40);
  logsoftmax40<<<(NN + 7) / 8, blk, 0, stream>>>(agg4, b4, out, NN);
}
